// IntegratedWTAFoveatedSNN_27616639713859
// MI455X (gfx1250) — matrix operand order repaired and checked by reading, not compiled, not run on hardware
//
#include <hip/hip_runtime.h>

typedef __attribute__((ext_vector_type(16))) __bf16 v16bf;
typedef __attribute__((ext_vector_type(8)))  float  v8f;

// ---- problem constants ----
constexpr int   kT = 32, kB = 32, kH = 128, kW = 128, kC = 2;
constexpr float kTAU = 4.0f, kBETA = 0.9f, kTHRESH = 1.0f;

// ---- output layout (float offsets) ----
constexpr size_t OUT_READOUT = 0;                   // 32*128   = 4096
constexpr size_t OUT_LOGITS  = 4096;                // 32*32*128= 131072
constexpr size_t OUT_SR      = OUT_LOGITS + 131072; // 2
constexpr size_t OUT_ROUTE   = OUT_SR + 2;          // 1024
constexpr size_t OUT_CHAN    = OUT_ROUTE + 1024;    // 128

// ---- workspace layout (byte offsets) ----
constexpr size_t WS_SCORES      = 0;                                   // T*B*1024 f32
constexpr size_t WS_CROP        = WS_SCORES + (size_t)kT*kB*1024*4;    // T*B int2
constexpr size_t WS_FUSED       = WS_CROP   + (size_t)kT*kB*8;         // T*B*192 f32
constexpr size_t WS_VF          = WS_FUSED  + (size_t)kT*kB*192*4;     // B*1024*128 f32
constexpr size_t WS_FUSED_BYTES = (size_t)kT*kB*192*4;

// -------- WMMA fragment loaders (CDNA5 16x16x32 bf16 layouts, wave32) --------
// A (16x32, 16-bit): lanes 0-15 -> M=lane, K 0..7 (v0-3) / 16..23 (v4-7);
//                    lanes16-31 -> M=lane-16, K 8..15 / 24..31.
// Per lane this is bytes [0,16)+[32,48) or [16,32)+[48,64) of a 64B row:
// compiler emits 2x ds_load_b128.
__device__ inline v16bf load_fragA(const __bf16* base, int ldk) {
  int l  = threadIdx.x & 31;
  int m  = l & 15;
  int kh = (l & 16) ? 8 : 0;
  v16bf r;
#pragma unroll
  for (int e = 0; e < 16; ++e) {
    int j = e >> 1, rb = e & 1;
    int k = (j < 4) ? (kh + 2*j + rb) : (16 + kh + 2*(j-4) + rb);
    r[e] = base[m*ldk + k];
  }
  return r;
}

// B fragments are staged in LDS already swizzled to per-lane-contiguous order:
// sBf[ntile][lane][0..15] (32 bytes/lane) -> one aligned v16bf load.
__device__ inline v16bf load_fragB_sw(const __bf16 (*sBf)[32][16], int n) {
  int l = threadIdx.x & 31;
  return *reinterpret_cast<const v16bf*>(&sBf[n][l][0]);
}

// Build swizzled B fragments from a KxN row-major f32 weight matrix
// (valid K rows = kvalid, rest zero-padded to K=32).
__device__ inline void build_B_frags(__bf16 (*sBf)[32][16], const float* W,
                                     int Ncols, int ntiles, int kvalid,
                                     int tid, int nthreads)
{
  for (int e = tid; e < ntiles*32*16; e += nthreads) {
    int n = e >> 9, l = (e >> 4) & 31, q = e & 15;
    int k = ((l & 16) ? 8 : 0) + q + ((q & 8) ? 8 : 0);
    int col = n*16 + (l & 15);
    float v = (k < kvalid) ? W[k*Ncols + col] : 0.0f;
    sBf[n][l][q] = (__bf16)v;
  }
}

// ------------------- kernel 1: patch scores, router, crop -------------------
__global__ __launch_bounds__(1024)
void k_scores(const float* __restrict__ x, float* __restrict__ scores_ws,
              int2* __restrict__ crop_ws, float* __restrict__ route_out)
{
  __shared__ float ls[1024];
  __shared__ int   best;
  const int tb = blockIdx.x, t = tb >> 5;
  const int p = threadIdx.x, py = p >> 5, px = p & 31;
  const float wt = __expf(-(float)(kT-1 - t) / kTAU);
  const float* xb = x + (size_t)tb * (kH*kW*kC);
  float s = 0.0f;
  for (int dy = 0; dy < 4; ++dy) {
    int y = py*4 + dy;
    for (int dx = 0; dx < 4; ++dx) {
      int xx = px*4 + dx;
      const float* pp = xb + (y*kW + xx)*kC;
      s += fabsf(pp[0]) + fabsf(pp[1]);
    }
  }
  s = wt * s * (1.0f/32.0f);
  ls[p] = s;
  if (p == 0) best = 0x7fffffff;
  __syncthreads();
  scores_ws[(size_t)tb*1024 + p] = s;
  int cnt = 0;
  for (int j = 0; j < 1024; ++j) cnt += (ls[j] > s) ? 1 : 0;
  if (cnt < 4) atomicAdd(route_out + p, 1.0f / (float)(kT*kB));
  if (cnt == 0) atomicMin(&best, p);
  __syncthreads();
  if (p == 0) {
    int bp = best;
    int cy = (bp >> 5)*4 + 2, cx = (bp & 31)*4 + 2;
    int sy = min(max(cy - 16, 0), kH - 32);
    int sx = min(max(cx - 16, 0), kW - 32);
    crop_ws[tb] = make_int2(sy, sx);
  }
}

// ---------------- kernel 2: fovea conv (WMMA) + kWTA + LIF scan ----------------
__device__ inline void build_fovea_A(__bf16 (*sA)[32], const float* xb,
                                     int2 cr, float wt, int hf, int tid)
{
  int row = tid >> 1, k0 = (tid & 1) * 16;
  int s = hf*512 + row, fy = s >> 5, fx = s & 31;
#pragma unroll
  for (int kk = 0; kk < 16; ++kk) {
    int k = k0 + kk;
    float v = 0.0f;
    if (k < 18) {
      int ky = k / 6, rem = k - ky*6, kx = rem >> 1, c = rem & 1;
      int ay = fy + ky - 1, ax = fx + kx - 1;
      if (ay >= 0 && ay < 32 && ax >= 0 && ax < 32)
        v = wt * xb[((cr.x + ay)*kW + (cr.y + ax))*kC + c];
    }
    sA[row][k] = (__bf16)v;
  }
}

__global__ __launch_bounds__(1024)
void k_fovea(const float* __restrict__ x, const float* __restrict__ fW,
             const float* __restrict__ fb, const int2* __restrict__ crop,
             float* __restrict__ Vf, float* __restrict__ fused,
             float* __restrict__ chan_out, float* __restrict__ sr_out)
{
  __shared__ __align__(32) __bf16 sA[512][32];     // 32 KB im2col half
  __shared__ __align__(32) __bf16 sBf[8][32][16];  //  8 KB swizzled B frags
  __shared__ float  sSal[128], sGate[128], sFus[128];
  __shared__ float  sTot;

  const int b = blockIdx.x, tid = threadIdx.x;
  const int wv = tid >> 5, lane = tid & 31;

  build_B_frags(sBf, fW, 128, 8, 18, tid, 1024);
  __syncthreads();

  float bias[8];
#pragma unroll
  for (int n = 0; n < 8; ++n) bias[n] = fb[n*16 + (lane & 15)];

  for (int t = 0; t < kT; ++t) {
    const int2  cr = crop[t*kB + b];
    const float wt = __expf(-(float)(kT-1 - t) / kTAU);
    const float* xb = x + (size_t)(t*kB + b) * (kH*kW*kC);

    if (tid < 128) { sSal[tid] = 0.0f; sFus[tid] = 0.0f; }
    if (tid == 0) sTot = 0.0f;
    __syncthreads();

    // pass 1: conv -> per-channel saliency
#pragma unroll
    for (int hf = 0; hf < 2; ++hf) {
      build_fovea_A(sA, xb, cr, wt, hf, tid);
      __syncthreads();
      v16bf a = load_fragA(&sA[wv*16][0], 32);
#pragma unroll
      for (int n = 0; n < 8; ++n) {
        v16bf bfr = load_fragB_sw(sBf, n);
        v8f c = {};
        c = __builtin_amdgcn_wmma_f32_16x16x32_bf16(false, a, false, bfr,
                                                    (short)0, c, false, false);
        float p = 0.0f;
#pragma unroll
        for (int i = 0; i < 8; ++i) p += fabsf(c[i] + bias[n]);
        p += __shfl_xor(p, 16, 32);                // combine M 0-7 with 8-15
        if (lane < 16) atomicAdd(&sSal[n*16 + lane], p);
      }
      __syncthreads();
    }

    // channel k-WTA (top-16 of 128) via rank
    if (tid < 128) {
      float my = sSal[tid];
      int cnt = 0;
      for (int j = 0; j < 128; ++j) cnt += (sSal[j] > my) ? 1 : 0;
      float g = (cnt < 16) ? 1.0f : 0.0f;
      sGate[tid] = g;
      if (g > 0.0f) atomicAdd(&chan_out[tid], 1.0f / (float)(kT*kB));
    }
    __syncthreads();

    // pass 2: conv -> gate -> LIF (Vf resident in L2 workspace)
    float myTot = 0.0f;
#pragma unroll
    for (int hf = 0; hf < 2; ++hf) {
      build_fovea_A(sA, xb, cr, wt, hf, tid);
      __syncthreads();
      v16bf a = load_fragA(&sA[wv*16][0], 32);
#pragma unroll
      for (int n = 0; n < 8; ++n) {
        v16bf bfr = load_fragB_sw(sBf, n);
        v8f c = {};
        c = __builtin_amdgcn_wmma_f32_16x16x32_bf16(false, a, false, bfr,
                                                    (short)0, c, false, false);
        int ch = n*16 + (lane & 15);
        float g = sGate[ch];
        float ssum = 0.0f;
#pragma unroll
        for (int i = 0; i < 8; ++i) {
          int m = i + ((lane & 16) ? 8 : 0);
          int srow = hf*512 + wv*16 + m;
          size_t vidx = ((size_t)b*1024 + srow)*128 + ch;
          float ff = (c[i] + bias[n]) * g;
          float v = (t == 0) ? 0.0f : Vf[vidx];
          v = kBETA*v + ff;
          float sp = (v > kTHRESH) ? 1.0f : 0.0f;
          Vf[vidx] = v - sp*kTHRESH;
          ssum += sp;
        }
        myTot += ssum;
        ssum += __shfl_xor(ssum, 16, 32);
        if (lane < 16) atomicAdd(&sFus[ch], ssum);
      }
      __syncthreads();
    }
    atomicAdd(&sTot, myTot);
    __syncthreads();
    if (tid < 128) fused[(size_t)(t*kB + b)*192 + tid] = sFus[tid] * (1.0f/1024.0f);
    if (tid == 0)
      atomicAdd(&sr_out[0], sTot / (float)((size_t)kT*kB*1024*128));
    __syncthreads();
  }
}

// ----- kernel 3: peripheral resize + conv (WMMA) + LIF scan (regs-resident) -----
__global__ __launch_bounds__(1024)
void k_periph(const float* __restrict__ x, const float* __restrict__ pW,
              const float* __restrict__ pb, float* __restrict__ fused,
              float* __restrict__ sr_out)
{
  __shared__ float  sP[18][66][2];                 // staged resized tile (+halo)
  __shared__ __align__(32) __bf16 sA[512][32];     // 32 KB im2col half
  __shared__ __align__(32) __bf16 sBf[4][32][16];  //  4 KB swizzled B frags
  __shared__ float  sSp[64];
  __shared__ float  sTot;

  const int b = blockIdx.x >> 2, q = blockIdx.x & 3;
  const int tid = threadIdx.x, wv = tid >> 5, lane = tid & 31;

  build_B_frags(sBf, pW, 64, 4, 18, tid, 1024);
  __syncthreads();

  float bias[4];
#pragma unroll
  for (int n = 0; n < 4; ++n) bias[n] = pb[n*16 + (lane & 15)];

  v8f zero = {};
  v8f vAcc[2][4];
#pragma unroll
  for (int h = 0; h < 2; ++h)
#pragma unroll
    for (int n = 0; n < 4; ++n) vAcc[h][n] = zero;

  for (int t = 0; t < kT; ++t) {
    const float wt = __expf(-(float)(kT-1 - t) / kTAU);
    const float* xb = x + (size_t)(t*kB + b) * (kH*kW*kC);

    if (tid < 64) sSp[tid] = 0.0f;
    if (tid == 0) sTot = 0.0f;
    __syncthreads();

    // antialiased 2x downsample, rows [q*16-1, q*16+17), cols [-1, 65)
    const float wk[4] = {0.125f, 0.375f, 0.375f, 0.125f};
    for (int e = tid; e < 18*66*2; e += 1024) {
      int l = e / 132, r2 = e - l*132, cc = r2 >> 1, ch = r2 & 1;
      int py = q*16 - 1 + l, px = cc - 1;
      float v = 0.0f;
      if (py >= 0 && py < 64 && px >= 0 && px < 64) {
#pragma unroll
        for (int i = 0; i < 4; ++i) {
          int iy = min(max(2*py - 1 + i, 0), kH - 1);
#pragma unroll
          for (int j = 0; j < 4; ++j) {
            int ix = min(max(2*px - 1 + j, 0), kW - 1);
            v += wk[i] * wk[j] * xb[(iy*kW + ix)*kC + ch];
          }
        }
        v *= wt;
      }
      sP[l][cc][ch] = v;
    }
    __syncthreads();

    float myTot = 0.0f;
#pragma unroll
    for (int hf = 0; hf < 2; ++hf) {
      {
        int row = tid >> 1, k0 = (tid & 1)*16;
        int s = hf*512 + row, fy = s >> 6, fx = s & 63;
#pragma unroll
        for (int kk = 0; kk < 16; ++kk) {
          int k = k0 + kk;
          float v = 0.0f;
          if (k < 18) {
            int ky = k / 6, rem = k - ky*6, kx = rem >> 1, ch = rem & 1;
            v = sP[fy + ky][fx + kx][ch];
          }
          sA[row][k] = (__bf16)v;
        }
      }
      __syncthreads();
      v16bf a = load_fragA(&sA[wv*16][0], 32);
#pragma unroll
      for (int n = 0; n < 4; ++n) {
        v16bf bfr = load_fragB_sw(sBf, n);
        v8f c = {};
        c = __builtin_amdgcn_wmma_f32_16x16x32_bf16(false, a, false, bfr,
                                                    (short)0, c, false, false);
        int ch = n*16 + (lane & 15);
        float ssum = 0.0f;
#pragma unroll
        for (int i = 0; i < 8; ++i) {
          float v = kBETA * vAcc[hf][n][i] + (c[i] + bias[n]);
          float sp = (v > kTHRESH) ? 1.0f : 0.0f;
          vAcc[hf][n][i] = v - sp*kTHRESH;
          ssum += sp;
        }
        myTot += ssum;
        ssum += __shfl_xor(ssum, 16, 32);
        if (lane < 16) atomicAdd(&sSp[ch], ssum);
      }
      __syncthreads();
    }
    atomicAdd(&sTot, myTot);
    __syncthreads();
    if (tid < 64)
      atomicAdd(&fused[(size_t)(t*kB + b)*192 + 128 + tid], sSp[tid] * (1.0f/4096.0f));
    if (tid == 0)
      atomicAdd(&sr_out[1], sTot / (float)((size_t)kT*kB*4096*64));
    __syncthreads();
  }
}

// ------------- kernel 4: logits = [fused ‖ scores] @ [head;route] + b -------------
__global__ __launch_bounds__(256)
void k_logits(const float* __restrict__ fused, const float* __restrict__ scores,
              const float* __restrict__ hW, const float* __restrict__ hb,
              const float* __restrict__ rW, const float* __restrict__ rb,
              float* __restrict__ out)
{
  __shared__ __align__(32) __bf16 sA[16][32];
  __shared__ __align__(32) __bf16 sBf[8][32][16];
  const int r0 = blockIdx.x * 16;
  const int wv = threadIdx.x >> 5, lane = threadIdx.x & 31;

  v8f acc = {};
  for (int chk = 0; chk < 38; ++chk) {   // 6 chunks fused(192) + 32 chunks scores(1024)
    for (int e = threadIdx.x; e < 512; e += 256) {
      int rr = e >> 5, kk = e & 31;
      float v = (chk < 6)
        ? fused [(size_t)(r0 + rr)*192  + chk*32 + kk]
        : scores[(size_t)(r0 + rr)*1024 + (chk - 6)*32 + kk];
      sA[rr][kk] = (__bf16)v;
    }
    // swizzled B fragments for this K-chunk (full 32 K-rows valid)
    {
      const float* W  = (chk < 6) ? (hW + chk*32*128) : (rW + (chk - 6)*32*128);
      build_B_frags(sBf, W, 128, 8, 32, threadIdx.x, 256);
    }
    __syncthreads();
    v16bf a   = load_fragA(&sA[0][0], 32);
    v16bf bfr = load_fragB_sw(sBf, wv);
    acc = __builtin_amdgcn_wmma_f32_16x16x32_bf16(false, a, false, bfr,
                                                  (short)0, acc, false, false);
    __syncthreads();
  }
  int ch = wv*16 + (lane & 15);
  float bias = hb[ch] + rb[ch];
#pragma unroll
  for (int i = 0; i < 8; ++i) {
    int m = i + ((lane & 16) ? 8 : 0);
    int r = r0 + m;
    float v = acc[i] + bias;
    out[OUT_LOGITS + (size_t)r*128 + ch] = v;           // logits_seq[t,b,:]
    atomicAdd(out + OUT_READOUT + (size_t)(r & 31)*128 + ch, v * (1.0f/32.0f));
  }
}

// ----------------------------------- launch -----------------------------------
extern "C" void kernel_launch(void* const* d_in, const int* in_sizes, int n_in,
                              void* d_out, int out_size, void* d_ws, size_t ws_size,
                              hipStream_t stream) {
  (void)in_sizes; (void)n_in; (void)ws_size;
  const float* x  = (const float*)d_in[0];
  const float* fW = (const float*)d_in[1];
  const float* fb = (const float*)d_in[2];
  const float* pW = (const float*)d_in[3];
  const float* pb = (const float*)d_in[4];
  const float* hW = (const float*)d_in[5];
  const float* hb = (const float*)d_in[6];
  const float* rW = (const float*)d_in[7];
  const float* rb = (const float*)d_in[8];
  float* out = (float*)d_out;

  char*  ws     = (char*)d_ws;
  float* scores = (float*)(ws + WS_SCORES);
  int2*  crop   = (int2*) (ws + WS_CROP);
  float* fused  = (float*)(ws + WS_FUSED);
  float* Vf     = (float*)(ws + WS_VF);

  hipMemsetAsync(d_out, 0, (size_t)out_size * sizeof(float), stream);
  hipMemsetAsync(fused, 0, WS_FUSED_BYTES, stream);

  k_scores<<<kT*kB, 1024, 0, stream>>>(x, scores, crop, out + OUT_ROUTE);
  k_fovea <<<kB,    1024, 0, stream>>>(x, fW, fb, crop, Vf, fused,
                                       out + OUT_CHAN, out + OUT_SR);
  k_periph<<<kB*4,  1024, 0, stream>>>(x, pW, pb, fused, out + OUT_SR);
  k_logits<<<64,     256, 0, stream>>>(fused, scores, hW, hb, rW, rb, out);
}
